// GCRM_412316860683
// MI455X (gfx1250) — matrix operand order repaired and checked by reading, not compiled, not run on hardware
//
#include <hip/hip_runtime.h>

#define Nn   2048
#define Ee   32768
#define DIN  32
#define Dd   64
#define EIN  16
#define DEe  64
#define Hh   4
#define Ll   3
#define Pp   5

typedef __attribute__((ext_vector_type(16))) __bf16 v16bf;
typedef __attribute__((ext_vector_type(8)))  __bf16 v8bf;
typedef __attribute__((ext_vector_type(8)))  float  v8f;

__device__ __forceinline__ v16bf cat16(v8bf lo, v8bf hi) {
  return __builtin_shufflevector(lo, hi, 0,1,2,3,4,5,6,7,8,9,10,11,12,13,14,15);
}

// ---------------- setup kernels ----------------

__global__ void k_zero_i32(int* p, int n) {
  int i = blockIdx.x * blockDim.x + threadIdx.x;
  if (i < n) p[i] = 0;
}

__global__ void k_deg(const int* __restrict__ ei, int* deg_out, int* deg_in) {
  int e = blockIdx.x * blockDim.x + threadIdx.x;
  if (e < Ee) {
    atomicAdd(&deg_out[ei[e]], 1);        // row 0 = source
    atomicAdd(&deg_in[ei[Ee + e]], 1);    // row 1 = target
  }
}

__global__ void k_h0(const float* __restrict__ x, const float* __restrict__ Wn,
                     const float* __restrict__ bn, const float* __restrict__ zin,
                     const float* __restrict__ zout, const float* __restrict__ zw,
                     const int* __restrict__ din, const int* __restrict__ dout,
                     const int* __restrict__ w, float* __restrict__ h) {
  int idx = blockIdx.x * blockDim.x + threadIdx.x;   // N*64
  int n = idx >> 6, c = idx & 63;
  float acc = bn[c];
  #pragma unroll
  for (int d = 0; d < DIN; ++d) acc += x[n * DIN + d] * Wn[d * Dd + c];
  int a  = min(din[n], 255);
  int b  = min(dout[n], 255);
  int ww = min(w[n], 99);
  h[idx] = acc + zin[a * Dd + c] + zout[b * Dd + c] + zw[ww * Dd + c];
}

__global__ void k_ea(const float* __restrict__ eattr, const float* __restrict__ We,
                     const float* __restrict__ be, float* __restrict__ ea) {
  int idx = blockIdx.x * blockDim.x + threadIdx.x;   // E*64
  int e = idx >> 6, c = idx & 63;
  float acc = be[c];
  #pragma unroll
  for (int d = 0; d < EIN; ++d) acc += eattr[e * EIN + d] * We[d * DEe + c];
  ea[idx] = acc;
}

// ---------------- per-layer kernels ----------------

// one wave per row; lane handles cols lane, lane+32
__global__ void __launch_bounds__(32) k_ln(const float* __restrict__ in,
                                           const float* __restrict__ g,
                                           const float* __restrict__ b,
                                           float* __restrict__ out) {
  int row = blockIdx.x, lane = threadIdx.x;
  float x0 = in[row * Dd + lane], x1 = in[row * Dd + 32 + lane];
  float s = x0 + x1;
  #pragma unroll
  for (int off = 1; off < 32; off <<= 1) s += __shfl_xor(s, off, 32);
  float mu = s * (1.0f / 64.0f);
  float d0 = x0 - mu, d1 = x1 - mu;
  float v = d0 * d0 + d1 * d1;
  #pragma unroll
  for (int off = 1; off < 32; off <<= 1) v += __shfl_xor(v, off, 32);
  float rstd = rsqrtf(v * (1.0f / 64.0f) + 1e-5f);
  out[row * Dd + lane]      = d0 * rstd * g[lane]      + b[lane];
  out[row * Dd + 32 + lane] = d1 * rstd * g[lane + 32] + b[lane + 32];
}

// all heads of one layer; V stored transposed [64][N] for WMMA B-fragments
__global__ void k_qkv(const float* __restrict__ hn,
                      const float* __restrict__ Wq, const float* __restrict__ bq,
                      const float* __restrict__ Wk, const float* __restrict__ bk,
                      const float* __restrict__ Wv, const float* __restrict__ bv,
                      __bf16* __restrict__ Qbf, __bf16* __restrict__ Kbf,
                      __bf16* __restrict__ Vtb) {
  int idx = blockIdx.x * blockDim.x + threadIdx.x;   // N*H*64
  int n  = idx / (Hh * Dd);
  int r  = idx % (Hh * Dd);
  int hd = r / Dd, c = r % Dd;
  float q = bq[hd * Dd + c], k = bk[hd * Dd + c], v = bv[hd * Dd + c];
  #pragma unroll
  for (int d = 0; d < Dd; ++d) {
    float hv = hn[n * Dd + d];
    q += hv * Wq[(hd * Dd + d) * Dd + c];
    k += hv * Wk[(hd * Dd + d) * Dd + c];
    v += hv * Wv[(hd * Dd + d) * Dd + c];
  }
  Qbf[((size_t)hd * Nn + n) * Dd + c] = (__bf16)q;
  Kbf[((size_t)hd * Nn + n) * Dd + c] = (__bf16)k;
  Vtb[((size_t)hd * Dd + c) * Nn + n] = (__bf16)v;
}

// M4[e][p][hd] = <ea[e], edge_vec[l,hd,p]>  -- all heads at once so the
// N^2*P path gather only has to run once per layer.
__global__ void k_M_all(const float* __restrict__ ea, const float* __restrict__ evec,
                        float* __restrict__ M4) {
  int idx = blockIdx.x * blockDim.x + threadIdx.x;   // E*P*H
  if (idx >= Ee * Pp * Hh) return;
  int e  = idx / (Pp * Hh);
  int r  = idx % (Pp * Hh);
  int p  = r / Hh, hd = r % Hh;
  const float* er = ea + (size_t)e * DEe;
  const float* vv = evec + ((size_t)hd * Pp + p) * DEe;
  float acc = 0.f;
  #pragma unroll
  for (int d = 0; d < DEe; ++d) acc += er[d] * vv[d];
  M4[((size_t)e * Pp + p) * Hh + hd] = acc;
}

// bias_all[hd][i][j] = spatial + path-mean edge encoding, all 4 heads per
// gather (index traffic amortized 4x; this kernel is the HBM bottleneck).
__global__ void k_bias_all(const int* __restrict__ nd, const int* __restrict__ ep,
                           const float* __restrict__ bs, const float* __restrict__ M4,
                           float* __restrict__ bias_all) {
  size_t idx = (size_t)blockIdx.x * blockDim.x + threadIdx.x;   // N*N
  int d = nd[idx];
  if (d > Pp) d = Pp;
  float bsv = (d > 0) ? bs[d - 1] : 0.f;
  float s0 = 0.f, s1 = 0.f, s2 = 0.f, s3 = 0.f;
  int cnt = 0;
  #pragma unroll
  for (int p = 0; p < Pp; ++p) {
    int e = ep[idx * Pp + p];
    if (e >= 0) {
      const float4 m4 = *(const float4*)(M4 + ((size_t)e * Pp + p) * Hh);
      s0 += m4.x; s1 += m4.y; s2 += m4.z; s3 += m4.w;
      cnt++;
    }
  }
  float invc = (cnt > 0) ? 1.0f / (float)cnt : 0.f;
  bias_all[0 * (size_t)Nn * Nn + idx] = bsv + s0 * invc;
  bias_all[1 * (size_t)Nn * Nn + idx] = bsv + s1 * invc;
  bias_all[2 * (size_t)Nn * Nn + idx] = bsv + s2 * invc;
  bias_all[3 * (size_t)Nn * Nn + idx] = bsv + s3 * invc;
}

// Fused flash attention, all heads batched via blockIdx.y.
// 1 wave per 16 query rows, bf16 WMMA. Layouts used here (k contiguous in A and B, not AMD's table order):
//   A 16x32 bf16 : lane<16 -> row=lane, K=0..15 contiguous; lane>=16 -> K=16..31
//   B 32x16 bf16 : lane<16 -> col=lane, K=0..15 contiguous; lane>=16 -> K=16..31
//   C/D 16x16 f32: elem r -> (M = r + 8*(lane>=16), N = lane&15)
__global__ void __launch_bounds__(32) k_attn(const __bf16* __restrict__ Qall,
                                             const __bf16* __restrict__ Kall,
                                             const __bf16* __restrict__ Vtall,
                                             const float* __restrict__ bias_all,
                                             float* __restrict__ ho) {
  __shared__ __align__(64) __bf16 Plds[16 * 32];
  const int hd = blockIdx.y;
  const __bf16* Q    = Qall    + (size_t)hd * Nn * Dd;
  const __bf16* K    = Kall    + (size_t)hd * Nn * Dd;
  const __bf16* Vt   = Vtall   + (size_t)hd * Dd * Nn;
  const float*  bias = bias_all + (size_t)hd * Nn * Nn;
  float*        out  = ho + hd * Dd;

  const int lane = threadIdx.x;
  const int lg = lane >> 4, ln = lane & 15;
  const int rowbase = blockIdx.x * 16;

  // Q A-fragments for K-chunks [0,32) and [32,64)
  const __bf16* qrow = Q + (size_t)(rowbase + ln) * Dd;
  v16bf a0 = *(const v16bf*)(qrow +      16 * lg);
  v16bf a1 = *(const v16bf*)(qrow + 32 + 16 * lg);

  float mrow[8], lrow[8];
  v8f o0 = {}, o1 = {}, o2 = {}, o3 = {};
  #pragma unroll
  for (int r = 0; r < 8; ++r) { mrow[r] = -1e30f; lrow[r] = 0.f; }

  for (int tile = 0; tile < Nn; tile += 32) {
    if (tile + 32 < Nn)
      __builtin_prefetch(K + (size_t)(tile + 32 + ln) * Dd, 0, 0);

    // ---- S = Q * K^T for 32 keys (two 16-key subtiles) ----
    v8f s[2];
    #pragma unroll
    for (int t = 0; t < 2; ++t) {
      const __bf16* krow = K + (size_t)(tile + 16 * t + ln) * Dd + 16 * lg;
      v16bf b0 = *(const v16bf*)(krow);
      v16bf b1 = *(const v16bf*)(krow + 32);
      v8f acc = {};
      acc = __builtin_amdgcn_wmma_f32_16x16x32_bf16(false, a0, false, b0, (short)0, acc, false, false);
      acc = __builtin_amdgcn_wmma_f32_16x16x32_bf16(false, a1, false, b1, (short)0, acc, false, false);
      s[t] = acc;
    }

    // ---- bias + online softmax (row = VGPR index, reduce across 16 lanes) ----
    float alpha[8];
    #pragma unroll
    for (int r = 0; r < 8; ++r) {
      size_t i = (size_t)(rowbase + 8 * lg + r);
      float v0 = s[0][r] * 0.125f + bias[i * Nn + tile + ln];
      float v1 = s[1][r] * 0.125f + bias[i * Nn + tile + 16 + ln];
      float mx = fmaxf(v0, v1);
      #pragma unroll
      for (int off = 1; off < 16; off <<= 1) mx = fmaxf(mx, __shfl_xor(mx, off, 32));
      float mnew = fmaxf(mrow[r], mx);
      alpha[r] = __expf(mrow[r] - mnew);
      mrow[r] = mnew;
      float p0 = __expf(v0 - mnew);
      float p1 = __expf(v1 - mnew);
      float ps = p0 + p1;
      #pragma unroll
      for (int off = 1; off < 16; off <<= 1) ps += __shfl_xor(ps, off, 32);
      lrow[r] = lrow[r] * alpha[r] + ps;
      s[0][r] = p0;
      s[1][r] = p1;
    }
    #pragma unroll
    for (int r = 0; r < 8; ++r) {
      o0[r] *= alpha[r]; o1[r] *= alpha[r]; o2[r] *= alpha[r]; o3[r] *= alpha[r];
    }

    // ---- C-layout P -> LDS -> A-layout fragment ----
    __syncthreads();
    #pragma unroll
    for (int r = 0; r < 8; ++r) {
      Plds[(8 * lg + r) * 32 + ln]      = (__bf16)s[0][r];
      Plds[(8 * lg + r) * 32 + 16 + ln] = (__bf16)s[1][r];
    }
    __syncthreads();
    const __bf16* prow = Plds + ln * 32;
    v16bf ap = *(const v16bf*)(prow + 16 * lg);

    // ---- O += P * V  (B-fragments contiguous from V^T [64][N]) ----
    v16bf bv0 = *(const v16bf*)(Vt + (size_t)(ln)      * Nn + tile + 16 * lg);
    v16bf bv1 = *(const v16bf*)(Vt + (size_t)(16 + ln) * Nn + tile + 16 * lg);
    v16bf bv2 = *(const v16bf*)(Vt + (size_t)(32 + ln) * Nn + tile + 16 * lg);
    v16bf bv3 = *(const v16bf*)(Vt + (size_t)(48 + ln) * Nn + tile + 16 * lg);
    o0 = __builtin_amdgcn_wmma_f32_16x16x32_bf16(false, ap, false, bv0, (short)0, o0, false, false);
    o1 = __builtin_amdgcn_wmma_f32_16x16x32_bf16(false, ap, false, bv1, (short)0, o1, false, false);
    o2 = __builtin_amdgcn_wmma_f32_16x16x32_bf16(false, ap, false, bv2, (short)0, o2, false, false);
    o3 = __builtin_amdgcn_wmma_f32_16x16x32_bf16(false, ap, false, bv3, (short)0, o3, false, false);
  }

  // ---- epilogue: normalize, store into head_out slice (stride H*D=256) ----
  #pragma unroll
  for (int r = 0; r < 8; ++r) {
    size_t i = (size_t)(rowbase + 8 * lg + r);
    float inv = 1.0f / lrow[r];
    out[i * (Hh * Dd) + ln]      = o0[r] * inv;
    out[i * (Hh * Dd) + 16 + ln] = o1[r] * inv;
    out[i * (Hh * Dd) + 32 + ln] = o2[r] * inv;
    out[i * (Hh * Dd) + 48 + ln] = o3[r] * inv;
  }
}

__global__ void k_oproj(const float* __restrict__ ho, const float* __restrict__ Wo,
                        const float* __restrict__ bo, const float* __restrict__ h,
                        float* __restrict__ h2) {
  int idx = blockIdx.x * blockDim.x + threadIdx.x;   // N*64
  int n = idx >> 6, c = idx & 63;
  float acc = bo[c];
  #pragma unroll 4
  for (int k = 0; k < Hh * Dd; ++k) acc += ho[(size_t)n * (Hh * Dd) + k] * Wo[k * Dd + c];
  h2[idx] = acc + h[idx];
}

__global__ void k_ff(const float* __restrict__ hn, const float* __restrict__ Wff,
                     const float* __restrict__ bff, const float* __restrict__ h2,
                     float* __restrict__ h) {
  int idx = blockIdx.x * blockDim.x + threadIdx.x;   // N*64
  int n = idx >> 6, c = idx & 63;
  float acc = bff[c];
  #pragma unroll
  for (int d = 0; d < Dd; ++d) acc += hn[n * Dd + d] * Wff[d * Dd + c];
  h[idx] = acc + h2[idx];
}

__global__ void k_final(const float* __restrict__ h, float* __restrict__ out) {
  int c = threadIdx.x;
  out[c] = h[(size_t)(Nn - 1) * Dd + c];
}

// ---------------- host orchestration ----------------

extern "C" void kernel_launch(void* const* d_in, const int* in_sizes, int n_in,
                              void* d_out, int out_size, void* d_ws, size_t ws_size,
                              hipStream_t stream) {
  (void)in_sizes; (void)n_in; (void)out_size; (void)ws_size;
  const float* x          = (const float*)d_in[0];
  const float* edge_attr  = (const float*)d_in[1];
  const int*   weights    = (const int*)d_in[2];
  const int*   edge_index = (const int*)d_in[3];
  const int*   node_dist  = (const int*)d_in[4];
  const int*   edge_paths = (const int*)d_in[5];
  const float* W_node = (const float*)d_in[6];
  const float* b_node = (const float*)d_in[7];
  const float* W_edge = (const float*)d_in[8];
  const float* b_edge = (const float*)d_in[9];
  const float* z_in   = (const float*)d_in[10];
  const float* z_out  = (const float*)d_in[11];
  const float* z_w    = (const float*)d_in[12];
  const float* b_spat = (const float*)d_in[13];
  const float* Wq = (const float*)d_in[14];
  const float* bq = (const float*)d_in[15];
  const float* Wk = (const float*)d_in[16];
  const float* bk = (const float*)d_in[17];
  const float* Wv = (const float*)d_in[18];
  const float* bv = (const float*)d_in[19];
  const float* edge_vec = (const float*)d_in[20];
  const float* Wo  = (const float*)d_in[21];
  const float* bo  = (const float*)d_in[22];
  const float* ln1g = (const float*)d_in[23];
  const float* ln1b = (const float*)d_in[24];
  const float* ln2g = (const float*)d_in[25];
  const float* ln2b = (const float*)d_in[26];
  const float* Wff = (const float*)d_in[27];
  const float* bff = (const float*)d_in[28];

  char* ws = (char*)d_ws;
  size_t off = 0;
  auto take = [&](size_t bytes) -> char* {
    char* p = ws + off;
    off += (bytes + 255) & ~(size_t)255;
    return p;
  };
  float*  h    = (float*)take((size_t)Nn * Dd * 4);
  float*  h2   = (float*)take((size_t)Nn * Dd * 4);
  float*  hn   = (float*)take((size_t)Nn * Dd * 4);
  int*    deg  = (int*)take((size_t)2 * Nn * 4);
  int*    deg_in  = deg;
  int*    deg_out = deg + Nn;
  float*  ea   = (float*)take((size_t)Ee * DEe * 4);
  float*  M4   = (float*)take((size_t)Ee * Pp * Hh * 4);
  float*  biasA= (float*)take((size_t)Hh * Nn * Nn * 4);
  __bf16* Qbf  = (__bf16*)take((size_t)Hh * Nn * Dd * 2);
  __bf16* Kbf  = (__bf16*)take((size_t)Hh * Nn * Dd * 2);
  __bf16* Vtb  = (__bf16*)take((size_t)Hh * Dd * Nn * 2);
  float*  ho   = (float*)take((size_t)Nn * Hh * Dd * 4);

  k_zero_i32<<<(2 * Nn + 255) / 256, 256, 0, stream>>>(deg, 2 * Nn);
  k_deg<<<(Ee + 255) / 256, 256, 0, stream>>>(edge_index, deg_out, deg_in);
  k_h0<<<(Nn * Dd) / 256, 256, 0, stream>>>(x, W_node, b_node, z_in, z_out, z_w,
                                            deg_in, deg_out, weights, h);
  k_ea<<<(Ee * DEe) / 256, 256, 0, stream>>>(edge_attr, W_edge, b_edge, ea);

  for (int l = 0; l < Ll; ++l) {
    k_ln<<<Nn, 32, 0, stream>>>(h, ln1g + l * Dd, ln1b + l * Dd, hn);
    k_qkv<<<(Nn * Hh * Dd) / 256, 256, 0, stream>>>(
        hn,
        Wq + (size_t)l * Hh * Dd * Dd, bq + (size_t)l * Hh * Dd,
        Wk + (size_t)l * Hh * Dd * Dd, bk + (size_t)l * Hh * Dd,
        Wv + (size_t)l * Hh * Dd * Dd, bv + (size_t)l * Hh * Dd,
        Qbf, Kbf, Vtb);
    k_M_all<<<(Ee * Pp * Hh + 255) / 256, 256, 0, stream>>>(
        ea, edge_vec + (size_t)l * Hh * Pp * DEe, M4);
    k_bias_all<<<(int)(((size_t)Nn * Nn) / 256), 256, 0, stream>>>(
        node_dist, edge_paths, b_spat, M4, biasA);
    k_attn<<<dim3(Nn / 16, Hh), 32, 0, stream>>>(Qbf, Kbf, Vtb, biasA, ho);
    k_oproj<<<(Nn * Dd) / 256, 256, 0, stream>>>(
        ho, Wo + (size_t)l * Hh * Dd * Dd, bo + (size_t)l * Dd, h, h2);
    k_ln<<<Nn, 32, 0, stream>>>(h2, ln2g + l * Dd, ln2b + l * Dd, hn);
    k_ff<<<(Nn * Dd) / 256, 256, 0, stream>>>(
        hn, Wff + (size_t)l * Dd * Dd, bff + (size_t)l * Dd, h2, h);
  }
  k_final<<<1, 64, 0, stream>>>(h, (float*)d_out);
}
